// SpatialAttention_61856118997496
// MI455X (gfx1250) — matrix operand order repaired and checked by reading, not compiled, not run on hardware
//
#include <hip/hip_runtime.h>
#include <hip/hip_bf16.h>

typedef __attribute__((ext_vector_type(16))) __bf16 bf16x16;
typedef __attribute__((ext_vector_type(8)))  float  floatx8;

union Frag16 { uint4 u[2]; bf16x16 v; };

__device__ __forceinline__ bf16x16 ld_bf16x16(const __bf16* p0, const __bf16* p1) {
    Frag16 f;
    f.u[0] = *reinterpret_cast<const uint4*>(p0);
    f.u[1] = *reinterpret_cast<const uint4*>(p1);
    return f.v;
}

__device__ __forceinline__ bf16x16 cvt_a_f32(const float* p0, const float* p1) {
    bf16x16 r;
#pragma unroll
    for (int e = 0; e < 8; ++e) r[e] = (__bf16)p0[e];
#pragma unroll
    for (int e = 0; e < 8; ++e) r[8 + e] = (__bf16)p1[e];
    return r;
}

__device__ __forceinline__ floatx8 wmma_bf16(bf16x16 a, bf16x16 b, floatx8 c) {
    return __builtin_amdgcn_wmma_f32_16x16x32_bf16(false, a, false, b, (short)0, c,
                                                   false, false);
}

__device__ __forceinline__ float rmax16(float x) {
    x = fmaxf(x, __shfl_xor(x, 1, 32));
    x = fmaxf(x, __shfl_xor(x, 2, 32));
    x = fmaxf(x, __shfl_xor(x, 4, 32));
    x = fmaxf(x, __shfl_xor(x, 8, 32));
    return x;
}
__device__ __forceinline__ float rsum16(float x) {
    x += __shfl_xor(x, 1, 32);
    x += __shfl_xor(x, 2, 32);
    x += __shfl_xor(x, 4, 32);
    x += __shfl_xor(x, 8, 32);
    return x;
}

#define NTOK   4096   // H*W
#define DMODEL 256
#define NROWS  16384  // B*NTOK
#define WSZ    (DMODEL * DMODEL)

// ---------------------------------------------------------------------------
// Tensor Data Mover: 2-D tile load (bf16 elements) via D# descriptor.
// clang-23 / therock lane: 6-arg builtin form.
// ---------------------------------------------------------------------------
#if __has_builtin(__builtin_amdgcn_tensor_load_to_lds) && \
    __has_builtin(__builtin_amdgcn_s_wait_tensorcnt)
#define HAVE_TDM 1
typedef __attribute__((ext_vector_type(4))) unsigned int u32x4;
typedef __attribute__((ext_vector_type(8))) int          i32x8;
typedef __attribute__((ext_vector_type(4))) int          i32x4;

// Loads a tile_d1 x tile_d0 (rows x contiguous cols) tile of a 2-D bf16 tensor
// whose row stride is stride0 elements, into LDS at lds_off (linear).
__device__ __forceinline__ void tdm_load_2d_bf16(unsigned lds_off, const void* gaddr,
                                                 unsigned tensor_d0, unsigned tensor_d1,
                                                 unsigned tile_d0, unsigned tile_d1,
                                                 unsigned stride0) {
    const unsigned long long ga = (unsigned long long)(size_t)gaddr;
    u32x4 g0;
    g0.x = 1u;                                                // count=1, user mode
    g0.y = lds_off;                                           // lds_addr
    g0.z = (unsigned)(ga & 0xFFFFFFFFu);                      // global_addr[31:0]
    g0.w = (unsigned)((ga >> 32) & 0x01FFFFFFu) | (2u << 30); // addr[56:32], type=2
    i32x8 g1;
    g1[0] = (int)(1u << 16);                                  // data_size = 2 bytes
    g1[1] = (int)((tensor_d0 & 0xFFFFu) << 16);               // tensor_dim0[15:0]
    g1[2] = (int)((tensor_d0 >> 16) | ((tensor_d1 & 0xFFFFu) << 16));
    g1[3] = (int)((tensor_d1 >> 16) | ((tile_d0 & 0xFFFFu) << 16));
    g1[4] = (int)(tile_d1 & 0xFFFFu);                         // tile_dim1, tile_dim2=0
    g1[5] = (int)stride0;                                     // tensor_dim0_stride[31:0]
    g1[6] = 0;
    g1[7] = 0;
    i32x4 zz4 = {0, 0, 0, 0};
    i32x8 zz8 = {0, 0, 0, 0, 0, 0, 0, 0};
    __builtin_amdgcn_tensor_load_to_lds(g0, g1, zz4, zz4, zz8, 0);
}
#endif

// ---------------------------------------------------------------------------
// Kernel 0: convert the 4 weight matrices to bf16, TRANSPOSED (Wt[n][k]).
// grid = (DMODEL*DMODEL/256, 4), block = 256.
// ---------------------------------------------------------------------------
__global__ __launch_bounds__(256) void prep_weights(
    const float* __restrict__ Wq, const float* __restrict__ Wk,
    const float* __restrict__ Wv, const float* __restrict__ Wo,
    __bf16* __restrict__ WtQ, __bf16* __restrict__ WtK,
    __bf16* __restrict__ WtV, __bf16* __restrict__ WtO)
{
    const int which = blockIdx.y;
    const float* W = (which == 0) ? Wq : (which == 1) ? Wk : (which == 2) ? Wv : Wo;
    __bf16* Wt     = (which == 0) ? WtQ : (which == 1) ? WtK : (which == 2) ? WtV : WtO;
    const int idx = blockIdx.x * 256 + threadIdx.x;
    const int k = idx >> 8;
    const int n = idx & 255;
    Wt[n * DMODEL + k] = (__bf16)W[idx];
}

// ---------------------------------------------------------------------------
// Kernel 1: fused QKV projection.  X:[16384,256] f32 x Wt(bf16,[n][k]).
// Q: row-major bf16 with 1/sqrt(D) folded.  K: row-major bf16.
// V: stored TRANSPOSED globally: Vt[b][d][n]  (so attention can TDM-load it).
// grid = (NROWS/64, DMODEL/64, 3), block = 128 (4 waves, 16 rows, 64 cols).
// ---------------------------------------------------------------------------
__global__ __launch_bounds__(128) void qkv_kernel(
    const float* __restrict__ X,
    const __bf16* __restrict__ WtQ, const __bf16* __restrict__ WtK,
    const __bf16* __restrict__ WtV,
    const float* __restrict__ bq, const float* __restrict__ bk,
    const float* __restrict__ bv,
    __bf16* __restrict__ Qb, __bf16* __restrict__ Kb, __bf16* __restrict__ VtG)
{
    const int which = blockIdx.z;
    const __bf16* Wt  = (which == 0) ? WtQ : (which == 1) ? WtK : WtV;
    const float* bias = (which == 0) ? bq : (which == 1) ? bk : bv;

    const int wave = threadIdx.x >> 5;
    const int lane = threadIdx.x & 31;
    const int g    = lane >> 4;
    const int l15  = lane & 15;

    const int row0 = blockIdx.x * 64 + wave * 16;
    const int col0 = blockIdx.y * 64;
    const float* Xrow = X + (size_t)(row0 + l15) * DMODEL;

    floatx8 acc[4];
    floatx8 z = {};
#pragma unroll
    for (int c = 0; c < 4; ++c) acc[c] = z;

#pragma unroll
    for (int kt = 0; kt < DMODEL; kt += 32) {
        bf16x16 a = cvt_a_f32(Xrow + kt + 16 * g, Xrow + kt + 16 * g + 8);
#pragma unroll
        for (int c = 0; c < 4; ++c) {
            const int col = col0 + c * 16 + l15;
            bf16x16 bm = ld_bf16x16(&Wt[(size_t)col * DMODEL + kt + 16 * g],
                                    &Wt[(size_t)col * DMODEL + kt + 16 * g + 8]);
            acc[c] = wmma_bf16(a, bm, acc[c]);
        }
    }

#pragma unroll
    for (int c = 0; c < 4; ++c) {
        const int col = col0 + c * 16 + l15;
        const float bb = bias[col];
        if (which == 2) {  // V -> transposed global layout Vt[b][d][n]
#pragma unroll
            for (int j = 0; j < 8; ++j) {
                const int orow = row0 + j + 8 * g;       // b*NTOK + n
                const int bb_i = orow >> 12;             // orow / 4096
                const int n    = orow & 4095;
                VtG[((size_t)bb_i * DMODEL + col) * NTOK + n] = (__bf16)(acc[c][j] + bb);
            }
        } else {
            __bf16* Out = (which == 0) ? Qb : Kb;
            const float oscale = (which == 0) ? 0.0625f : 1.0f;
#pragma unroll
            for (int j = 0; j < 8; ++j) {
                const int orow = row0 + j + 8 * g;
                Out[(size_t)orow * DMODEL + col] = (__bf16)((acc[c][j] + bb) * oscale);
            }
        }
    }
}

// ---------------------------------------------------------------------------
// Kernel 2: flash attention.  grid = (NTOK/64, B), block = 128 (4 waves).
// Each wave owns 16 q-rows; 32-key steps.  K / V^T tiles are DOUBLE-BUFFERED
// in LDS and streamed by the Tensor Data Mover one step ahead, so the DMA for
// step i+1 overlaps the WMMAs of step i (TENSORcnt completes in-order, so
// s_wait_tensorcnt 2 after issuing the next 2 loads means step i's tiles are
// resident).  Fallback: vectorized copies.
// ---------------------------------------------------------------------------
__global__ __launch_bounds__(128) void attn_kernel(
    const __bf16* __restrict__ Qb, const __bf16* __restrict__ Kb,
    const __bf16* __restrict__ VtG, __bf16* __restrict__ Ob)
{
    __shared__ __align__(16) __bf16 kTile[2][32 * DMODEL];   // [buf][key][d]
    __shared__ __align__(16) __bf16 vtTile[2][DMODEL * 32];  // [buf][d][key]
    __shared__ __align__(16) __bf16 pTile[4][16 * 32];       // per-wave P scratch

    const int b    = blockIdx.y;
    const int tid  = threadIdx.x;
    const int wave = tid >> 5;
    const int lane = tid & 31;
    const int g    = lane >> 4;
    const int l15  = lane & 15;

    const int qrow0 = blockIdx.x * 64 + wave * 16;
    const __bf16* Qrow = Qb + (size_t)(b * NTOK + qrow0 + l15) * DMODEL;

    // Q fragments held in registers: 8 chunks of 32 along d.
    bf16x16 qf[8];
#pragma unroll
    for (int c = 0; c < 8; ++c)
        qf[c] = ld_bf16x16(Qrow + c * 32 + 16 * g, Qrow + c * 32 + 16 * g + 8);

    floatx8 acc[16];
    floatx8 z = {};
#pragma unroll
    for (int c = 0; c < 16; ++c) acc[c] = z;
    float m[8], l[8];
#pragma unroll
    for (int j = 0; j < 8; ++j) { m[j] = -__builtin_inff(); l[j] = 0.0f; }

    const int NSTEPS = NTOK / 32;
    const size_t kmat0 = (size_t)b * NTOK * DMODEL;   // K base for this batch
    const size_t vmat0 = (size_t)b * DMODEL * NTOK;   // Vt base for this batch

#ifdef HAVE_TDM
    if (wave == 0) {
        // Prefetch step 0 into buffer 0.
        tdm_load_2d_bf16((unsigned)(size_t)&kTile[0][0], &Kb[kmat0],
                         DMODEL, NTOK, DMODEL, 32, DMODEL);
        tdm_load_2d_bf16((unsigned)(size_t)&vtTile[0][0], &VtG[vmat0],
                         NTOK, DMODEL, 32, DMODEL, NTOK);
    }
#endif

    for (int i = 0; i < NSTEPS; ++i) {
        const int buf = i & 1;
        __syncthreads();  // step i-1 fully consumed -> buf^1 free for prefetch
#ifdef HAVE_TDM
        if (wave == 0) {
            if (i + 1 < NSTEPS) {
                const int kn = (i + 1) * 32;
                tdm_load_2d_bf16((unsigned)(size_t)&kTile[buf ^ 1][0],
                                 &Kb[kmat0 + (size_t)kn * DMODEL],
                                 DMODEL, NTOK, DMODEL, 32, DMODEL);
                tdm_load_2d_bf16((unsigned)(size_t)&vtTile[buf ^ 1][0],
                                 &VtG[vmat0 + kn],
                                 NTOK, DMODEL, 32, DMODEL, NTOK);
                __builtin_amdgcn_s_wait_tensorcnt(2);  // step i's 2 loads done
            } else {
                __builtin_amdgcn_s_wait_tensorcnt(0);
            }
        }
#else
        {
            const int kb0 = i * 32;
            const size_t kbase = kmat0 + (size_t)kb0 * DMODEL;
            for (int idx = tid; idx < (32 * DMODEL) / 8; idx += 128) {
                const int off = idx * 8;
                *reinterpret_cast<uint4*>(&kTile[buf][off]) =
                    *reinterpret_cast<const uint4*>(&Kb[kbase + off]);
            }
            for (int idx = tid; idx < (32 * DMODEL) / 8; idx += 128) {
                const int off = idx * 8;
                const int d = off >> 5, c = off & 31;
                *reinterpret_cast<uint4*>(&vtTile[buf][off]) =
                    *reinterpret_cast<const uint4*>(
                        &VtG[vmat0 + (size_t)d * NTOK + kb0 + c]);
            }
        }
#endif
        __syncthreads();  // tiles for step i visible to all waves

        // ---- S = Q K^T for 32 keys (two 16x16 f32 fragments) ----
        floatx8 sA = z, sB = z;
#pragma unroll
        for (int c = 0; c < 8; ++c) {
            bf16x16 kfA = ld_bf16x16(&kTile[buf][l15 * DMODEL + c * 32 + 16 * g],
                                     &kTile[buf][l15 * DMODEL + c * 32 + 16 * g + 8]);
            sA = wmma_bf16(qf[c], kfA, sA);
            bf16x16 kfB = ld_bf16x16(&kTile[buf][(16 + l15) * DMODEL + c * 32 + 16 * g],
                                     &kTile[buf][(16 + l15) * DMODEL + c * 32 + 16 * g + 8]);
            sB = wmma_bf16(qf[c], kfB, sB);
        }

        // ---- online softmax update ----
        float alpha[8];
#pragma unroll
        for (int j = 0; j < 8; ++j) {
            float mb = rmax16(fmaxf(sA[j], sB[j]));
            float mn = fmaxf(m[j], mb);
            float a  = __expf(m[j] - mn);
            float pA = __expf(sA[j] - mn);
            float pB = __expf(sB[j] - mn);
            float rs = rsum16(pA + pB);
            l[j] = l[j] * a + rs;
            m[j] = mn;
            alpha[j] = a;
            pTile[wave][(j + 8 * g) * 32 + l15]      = (__bf16)pA;
            pTile[wave][(j + 8 * g) * 32 + 16 + l15] = (__bf16)pB;
        }
#pragma unroll
        for (int c = 0; c < 16; ++c)
#pragma unroll
            for (int j = 0; j < 8; ++j) acc[c][j] *= alpha[j];

        asm volatile("s_wait_dscnt 0" ::: "memory");  // wave-private LDS RAW

        bf16x16 pf = ld_bf16x16(&pTile[wave][l15 * 32 + 16 * g],
                                &pTile[wave][l15 * 32 + 16 * g + 8]);

        // ---- acc += P * V (16 d-chunks) ----
#pragma unroll
        for (int c = 0; c < 16; ++c) {
            bf16x16 vf = ld_bf16x16(&vtTile[buf][(c * 16 + l15) * 32 + 16 * g],
                                    &vtTile[buf][(c * 16 + l15) * 32 + 16 * g + 8]);
            acc[c] = wmma_bf16(pf, vf, acc[c]);
        }
    }

    float invl[8];
#pragma unroll
    for (int j = 0; j < 8; ++j) invl[j] = 1.0f / l[j];
#pragma unroll
    for (int c = 0; c < 16; ++c) {
        const int col = c * 16 + l15;
#pragma unroll
        for (int j = 0; j < 8; ++j) {
            const int row = qrow0 + j + 8 * g;
            Ob[(size_t)(b * NTOK + row) * DMODEL + col] = (__bf16)(acc[c][j] * invl[j]);
        }
    }
}

// ---------------------------------------------------------------------------
// Kernel 3: output projection.  Ob bf16 [16384,256] x WtO(bf16,[n][k]) + bo
// -> f32 out.  grid = (NROWS/64, DMODEL/64), block = 128.
// ---------------------------------------------------------------------------
__global__ __launch_bounds__(128) void out_proj(
    const __bf16* __restrict__ Ob, const __bf16* __restrict__ WtO,
    const float* __restrict__ bo, float* __restrict__ Out)
{
    const int wave = threadIdx.x >> 5;
    const int lane = threadIdx.x & 31;
    const int g    = lane >> 4;
    const int l15  = lane & 15;

    const int row0 = blockIdx.x * 64 + wave * 16;
    const int col0 = blockIdx.y * 64;
    const __bf16* Arow = Ob + (size_t)(row0 + l15) * DMODEL;

    floatx8 acc[4];
    floatx8 z = {};
#pragma unroll
    for (int c = 0; c < 4; ++c) acc[c] = z;

#pragma unroll
    for (int kt = 0; kt < DMODEL; kt += 32) {
        bf16x16 a = ld_bf16x16(Arow + kt + 16 * g, Arow + kt + 16 * g + 8);
#pragma unroll
        for (int c = 0; c < 4; ++c) {
            const int col = col0 + c * 16 + l15;
            bf16x16 bm = ld_bf16x16(&WtO[(size_t)col * DMODEL + kt + 16 * g],
                                    &WtO[(size_t)col * DMODEL + kt + 16 * g + 8]);
            acc[c] = wmma_bf16(a, bm, acc[c]);
        }
    }

#pragma unroll
    for (int c = 0; c < 4; ++c) {
        const int col = col0 + c * 16 + l15;
        const float bb = bo[col];
#pragma unroll
        for (int j = 0; j < 8; ++j) {
            const int orow = row0 + j + 8 * g;
            Out[(size_t)orow * DMODEL + col] = acc[c][j] + bb;
        }
    }
}

// ---------------------------------------------------------------------------
extern "C" void kernel_launch(void* const* d_in, const int* in_sizes, int n_in,
                              void* d_out, int out_size, void* d_ws, size_t ws_size,
                              hipStream_t stream) {
    const float* X  = (const float*)d_in[0];
    const float* Wq = (const float*)d_in[1];
    const float* bq = (const float*)d_in[2];
    const float* Wk = (const float*)d_in[3];
    const float* bk = (const float*)d_in[4];
    const float* Wv = (const float*)d_in[5];
    const float* bv = (const float*)d_in[6];
    const float* Wo = (const float*)d_in[7];
    const float* bo = (const float*)d_in[8];
    float* Out = (float*)d_out;

    const size_t SZ = (size_t)NROWS * DMODEL;  // elements per bf16 matrix
    __bf16* base = (__bf16*)d_ws;
    __bf16* Qb  = base;
    __bf16* Kb  = base + SZ;
    __bf16* VtG = base + 2 * SZ;
    __bf16* Ob  = base + 3 * SZ;
    __bf16* WtQ = base + 4 * SZ;
    __bf16* WtK = WtQ + WSZ;
    __bf16* WtV = WtK + WSZ;
    __bf16* WtO = WtV + WSZ;

    dim3 g0(WSZ / 256, 4);
    prep_weights<<<g0, 256, 0, stream>>>(Wq, Wk, Wv, Wo, WtQ, WtK, WtV, WtO);

    dim3 g1(NROWS / 64, DMODEL / 64, 3);
    qkv_kernel<<<g1, 128, 0, stream>>>(X, WtQ, WtK, WtV, bq, bk, bv, Qb, Kb, VtG);

    dim3 g2(NTOK / 64, 4);
    attn_kernel<<<g2, 128, 0, stream>>>(Qb, Kb, VtG, Ob);

    dim3 g3(NROWS / 64, DMODEL / 64);
    out_proj<<<g3, 128, 0, stream>>>(Ob, WtO, bo, Out);
}
